// ThorMoE_15564961481511
// MI455X (gfx1250) — matrix operand order repaired and checked by reading, not compiled, not run on hardware
//
#include <hip/hip_runtime.h>

#define HDIM 1024
#define IDIM 4096
#define NEXP 8
#define CAP  512
#define TTOT 4096

typedef __attribute__((ext_vector_type(16))) __bf16    v16bf;
typedef __attribute__((ext_vector_type(8)))  float     v8f;
typedef __attribute__((ext_vector_type(4)))  float     f4;
typedef __attribute__((ext_vector_type(4)))  unsigned  u4;

union Frag { v16bf v; unsigned u[8]; };

// fp32 -> bf16 round-to-nearest-even, pack two into one dword
__device__ __forceinline__ unsigned pk_bf16(float lo, float hi) {
    unsigned a = __float_as_uint(lo), b = __float_as_uint(hi);
    a = (a + 0x7FFFu + ((a >> 16) & 1u)) >> 16;
    b = (b + 0x7FFFu + ((b >> 16) & 1u)) >> 16;
    return (b << 16) | (a & 0xFFFFu);
}
__device__ __forceinline__ unsigned short cvt_bf16(float x) {
    unsigned a = __float_as_uint(x);
    return (unsigned short)((a + 0x7FFFu + ((a >> 16) & 1u)) >> 16);
}

#define A_STRIDE 17          // padded dword stride: 17 coprime w/ 64 banks
#define B_STRIDE 260         // 256 + 4 pad: breaks lane-half bank aliasing
#define A_BUF    (128 * A_STRIDE)
#define B_BUF    (16 * B_STRIDE)

// ---------------------------------------------------------------------------
// GEMM1: h[e,c,i] = bf16( gather(x,perm)[e,c,:] @ W1[e] + b1[e] )
// block tile 128(M) x 256(N), K-step 32, 8 waves (2x4), wave tile 64x64,
// double-buffered LDS, NT weight streaming.
// ---------------------------------------------------------------------------
__global__ __launch_bounds__(256)
void moe_gemm1(const float* __restrict__ x, const int* __restrict__ perm,
               const float* __restrict__ W1, const float* __restrict__ b1,
               unsigned short* __restrict__ hbuf) {
    __shared__ unsigned As[2 * A_BUF];
    __shared__ unsigned Bs[2 * B_BUF];

    const int e   = blockIdx.z;
    const int m0  = blockIdx.y * 128;
    const int n0  = blockIdx.x * 256;
    const int tid = threadIdx.x;
    const float* W1e = W1 + (size_t)e * HDIM * IDIM;

    // per-thread gathered A-row pointers + LDS slots (fixed over K loop)
    const float* arow[4]; int ar[4], aq[4];
#pragma unroll
    for (int it = 0; it < 4; ++it) {
        int idx = tid + 256 * it;              // 128 rows x 8 float4-quads
        ar[it] = idx >> 3;  aq[it] = idx & 7;
        int tok = perm[e * CAP + m0 + ar[it]];
        arow[it] = x + (size_t)tok * HDIM + 4 * aq[it];
    }
    const float* brow[4]; int bp[4], bq[4];
#pragma unroll
    for (int it = 0; it < 4; ++it) {
        int idx = tid + 256 * it;              // 16 K-pairs x 64 quad-cols
        bp[it] = idx >> 6;  bq[it] = idx & 63;
        brow[it] = W1e + (size_t)(2 * bp[it]) * IDIM + n0 + 4 * bq[it];
    }

    const int lane  = tid & 31;
    const int kh    = lane >> 4;
    const int ln    = lane & 15;
    const int wave  = tid >> 5;
    const int mbase = (wave >> 2) * 64;
    const int nbase = (wave & 3) * 64;

    f4 fa[4], fb0[4], fb1[4];
    // ---- prologue: load K-step 0, fill buffer 0 ----
#pragma unroll
    for (int it = 0; it < 4; ++it) fa[it] = *(const f4*)(arow[it]);
#pragma unroll
    for (int it = 0; it < 4; ++it) {
        fb0[it] = __builtin_nontemporal_load((const f4*)brow[it]);
        fb1[it] = __builtin_nontemporal_load((const f4*)(brow[it] + IDIM));
    }
#pragma unroll
    for (int it = 0; it < 4; ++it) {
        As[ar[it] * A_STRIDE + 2 * aq[it] + 0] = pk_bf16(fa[it].x, fa[it].y);
        As[ar[it] * A_STRIDE + 2 * aq[it] + 1] = pk_bf16(fa[it].z, fa[it].w);
    }
#pragma unroll
    for (int it = 0; it < 4; ++it) {
        Bs[bp[it] * B_STRIDE + 4 * bq[it] + 0] = pk_bf16(fb0[it].x, fb1[it].x);
        Bs[bp[it] * B_STRIDE + 4 * bq[it] + 1] = pk_bf16(fb0[it].y, fb1[it].y);
        Bs[bp[it] * B_STRIDE + 4 * bq[it] + 2] = pk_bf16(fb0[it].z, fb1[it].z);
        Bs[bp[it] * B_STRIDE + 4 * bq[it] + 3] = pk_bf16(fb0[it].w, fb1[it].w);
    }

    v8f acc[4][4] = {};
    const int NK = HDIM / 32;
    for (int ks = 0; ks < NK; ++ks) {
        __syncthreads();                       // buffer ks&1 ready for all
        // ---- issue next step's global loads (overlap with WMMA) ----
        if (ks + 1 < NK) {
            const int kn = (ks + 1) * 32;
#pragma unroll
            for (int it = 0; it < 4; ++it) fa[it] = *(const f4*)(arow[it] + kn);
#pragma unroll
            for (int it = 0; it < 4; ++it) {
                const float* s0 = brow[it] + (size_t)kn * IDIM;
                fb0[it] = __builtin_nontemporal_load((const f4*)s0);
                fb1[it] = __builtin_nontemporal_load((const f4*)(s0 + IDIM));
            }
        }
        // ---- compute from current buffer ----
        const unsigned* Ab = As + (ks & 1) * A_BUF;
        const unsigned* Bb = Bs + (ks & 1) * B_BUF;
        Frag a[4], b[4];
#pragma unroll
        for (int mi = 0; mi < 4; ++mi) {
            int row = mbase + mi * 16 + ln;
#pragma unroll
            for (int i = 0; i < 4; ++i) {
                a[mi].u[i]     = Ab[row * A_STRIDE + 4 * kh + i];
                a[mi].u[4 + i] = Ab[row * A_STRIDE + 8 + 4 * kh + i];
            }
        }
#pragma unroll
        for (int ni = 0; ni < 4; ++ni) {
            int col = nbase + ni * 16 + ln;
#pragma unroll
            for (int i = 0; i < 8; ++i)
                b[ni].u[i] = Bb[(4 * kh + i + (i & 4)) * B_STRIDE + col];
        }
#pragma unroll
        for (int mi = 0; mi < 4; ++mi)
#pragma unroll
            for (int ni = 0; ni < 4; ++ni)
                acc[mi][ni] = __builtin_amdgcn_wmma_f32_16x16x32_bf16(
                    false, a[mi].v, false, b[ni].v, (short)0, acc[mi][ni],
                    false, false);
        // ---- fill alternate buffer for next step ----
        if (ks + 1 < NK) {
            unsigned* Aw = As + ((ks + 1) & 1) * A_BUF;
            unsigned* Bw = Bs + ((ks + 1) & 1) * B_BUF;
#pragma unroll
            for (int it = 0; it < 4; ++it) {
                Aw[ar[it] * A_STRIDE + 2 * aq[it] + 0] = pk_bf16(fa[it].x, fa[it].y);
                Aw[ar[it] * A_STRIDE + 2 * aq[it] + 1] = pk_bf16(fa[it].z, fa[it].w);
            }
#pragma unroll
            for (int it = 0; it < 4; ++it) {
                Bw[bp[it] * B_STRIDE + 4 * bq[it] + 0] = pk_bf16(fb0[it].x, fb1[it].x);
                Bw[bp[it] * B_STRIDE + 4 * bq[it] + 1] = pk_bf16(fb0[it].y, fb1[it].y);
                Bw[bp[it] * B_STRIDE + 4 * bq[it] + 2] = pk_bf16(fb0[it].z, fb1[it].z);
                Bw[bp[it] * B_STRIDE + 4 * bq[it] + 3] = pk_bf16(fb0[it].w, fb1[it].w);
            }
        }
    }

    // ---- epilogue: +b1, bf16, store h tile (RT stores: keep h in L2) ----
    float bias[4];
#pragma unroll
    for (int ni = 0; ni < 4; ++ni)
        bias[ni] = b1[(size_t)e * IDIM + n0 + nbase + ni * 16 + ln];

#pragma unroll
    for (int mi = 0; mi < 4; ++mi)
#pragma unroll
        for (int ni = 0; ni < 4; ++ni) {
            int n = n0 + nbase + ni * 16 + ln;
#pragma unroll
            for (int j = 0; j < 8; ++j) {
                int m   = m0 + mbase + mi * 16 + j + 8 * kh;
                float v = acc[mi][ni][j] + bias[ni];
                hbuf[((size_t)e * CAP + m) * IDIM + n] = cvt_bf16(v);
            }
        }
}

// ---------------------------------------------------------------------------
// GEMM2: out[perm[e*C+c], :] = h[e,c,:] @ W2[e] + b2[e]
// ---------------------------------------------------------------------------
__global__ __launch_bounds__(256)
void moe_gemm2(const unsigned* __restrict__ hbuf, const int* __restrict__ perm,
               const float* __restrict__ W2, const float* __restrict__ b2,
               float* __restrict__ out) {
    __shared__ unsigned As[2 * A_BUF];
    __shared__ unsigned Bs[2 * B_BUF];
    __shared__ int toks[128];

    const int e   = blockIdx.z;
    const int m0  = blockIdx.y * 128;
    const int n0  = blockIdx.x * 256;
    const int tid = threadIdx.x;
    const float* W2e = W2 + (size_t)e * IDIM * HDIM;

    if (tid < 128) toks[tid] = perm[e * CAP + m0 + tid];

    // h rows already bf16-pair packed: u4 = 4 dwords = 8 bf16
    const unsigned* hrow[2]; int ar[2], ac[2];
#pragma unroll
    for (int it = 0; it < 2; ++it) {
        int idx = tid + 256 * it;              // 128 rows x 4 u4-chunks
        ar[it] = idx >> 2;  ac[it] = idx & 3;
        hrow[it] = hbuf + ((size_t)e * CAP + m0 + ar[it]) * (IDIM / 2) + 4 * ac[it];
    }
    const float* brow[4]; int bp[4], bq[4];
#pragma unroll
    for (int it = 0; it < 4; ++it) {
        int idx = tid + 256 * it;
        bp[it] = idx >> 6;  bq[it] = idx & 63;
        brow[it] = W2e + (size_t)(2 * bp[it]) * HDIM + n0 + 4 * bq[it];
    }

    const int lane  = tid & 31;
    const int kh    = lane >> 4;
    const int ln    = lane & 15;
    const int wave  = tid >> 5;
    const int mbase = (wave >> 2) * 64;
    const int nbase = (wave & 3) * 64;

    u4 ha[2]; f4 fb0[4], fb1[4];
    // ---- prologue ----
#pragma unroll
    for (int it = 0; it < 2; ++it) ha[it] = *(const u4*)hrow[it];
#pragma unroll
    for (int it = 0; it < 4; ++it) {
        fb0[it] = __builtin_nontemporal_load((const f4*)brow[it]);
        fb1[it] = __builtin_nontemporal_load((const f4*)(brow[it] + HDIM));
    }
#pragma unroll
    for (int it = 0; it < 2; ++it) {
        As[ar[it] * A_STRIDE + 4 * ac[it] + 0] = ha[it].x;
        As[ar[it] * A_STRIDE + 4 * ac[it] + 1] = ha[it].y;
        As[ar[it] * A_STRIDE + 4 * ac[it] + 2] = ha[it].z;
        As[ar[it] * A_STRIDE + 4 * ac[it] + 3] = ha[it].w;
    }
#pragma unroll
    for (int it = 0; it < 4; ++it) {
        Bs[bp[it] * B_STRIDE + 4 * bq[it] + 0] = pk_bf16(fb0[it].x, fb1[it].x);
        Bs[bp[it] * B_STRIDE + 4 * bq[it] + 1] = pk_bf16(fb0[it].y, fb1[it].y);
        Bs[bp[it] * B_STRIDE + 4 * bq[it] + 2] = pk_bf16(fb0[it].z, fb1[it].z);
        Bs[bp[it] * B_STRIDE + 4 * bq[it] + 3] = pk_bf16(fb0[it].w, fb1[it].w);
    }

    v8f acc[4][4] = {};
    const int NK = IDIM / 32;
    for (int ks = 0; ks < NK; ++ks) {
        __syncthreads();
        if (ks + 1 < NK) {
            const int kn = (ks + 1) * 32;
#pragma unroll
            for (int it = 0; it < 2; ++it) ha[it] = *(const u4*)(hrow[it] + (kn >> 1));
#pragma unroll
            for (int it = 0; it < 4; ++it) {
                const float* s0 = brow[it] + (size_t)kn * HDIM;
                fb0[it] = __builtin_nontemporal_load((const f4*)s0);
                fb1[it] = __builtin_nontemporal_load((const f4*)(s0 + HDIM));
            }
        }
        const unsigned* Ab = As + (ks & 1) * A_BUF;
        const unsigned* Bb = Bs + (ks & 1) * B_BUF;
        Frag a[4], b[4];
#pragma unroll
        for (int mi = 0; mi < 4; ++mi) {
            int row = mbase + mi * 16 + ln;
#pragma unroll
            for (int i = 0; i < 4; ++i) {
                a[mi].u[i]     = Ab[row * A_STRIDE + 4 * kh + i];
                a[mi].u[4 + i] = Ab[row * A_STRIDE + 8 + 4 * kh + i];
            }
        }
#pragma unroll
        for (int ni = 0; ni < 4; ++ni) {
            int col = nbase + ni * 16 + ln;
#pragma unroll
            for (int i = 0; i < 8; ++i)
                b[ni].u[i] = Bb[(4 * kh + i + (i & 4)) * B_STRIDE + col];
        }
#pragma unroll
        for (int mi = 0; mi < 4; ++mi)
#pragma unroll
            for (int ni = 0; ni < 4; ++ni)
                acc[mi][ni] = __builtin_amdgcn_wmma_f32_16x16x32_bf16(
                    false, a[mi].v, false, b[ni].v, (short)0, acc[mi][ni],
                    false, false);
        if (ks + 1 < NK) {
            unsigned* Aw = As + ((ks + 1) & 1) * A_BUF;
            unsigned* Bw = Bs + ((ks + 1) & 1) * B_BUF;
#pragma unroll
            for (int it = 0; it < 2; ++it) {
                Aw[ar[it] * A_STRIDE + 4 * ac[it] + 0] = ha[it].x;
                Aw[ar[it] * A_STRIDE + 4 * ac[it] + 1] = ha[it].y;
                Aw[ar[it] * A_STRIDE + 4 * ac[it] + 2] = ha[it].z;
                Aw[ar[it] * A_STRIDE + 4 * ac[it] + 3] = ha[it].w;
            }
#pragma unroll
            for (int it = 0; it < 4; ++it) {
                Bw[bp[it] * B_STRIDE + 4 * bq[it] + 0] = pk_bf16(fb0[it].x, fb1[it].x);
                Bw[bp[it] * B_STRIDE + 4 * bq[it] + 1] = pk_bf16(fb0[it].y, fb1[it].y);
                Bw[bp[it] * B_STRIDE + 4 * bq[it] + 2] = pk_bf16(fb0[it].z, fb1[it].z);
                Bw[bp[it] * B_STRIDE + 4 * bq[it] + 3] = pk_bf16(fb0[it].w, fb1[it].w);
            }
        }
    }

    // ---- epilogue: +b2, scatter fp32 rows (NT stores: out never re-read) ----
    float bias[4];
#pragma unroll
    for (int ni = 0; ni < 4; ++ni)
        bias[ni] = b2[(size_t)e * HDIM + n0 + nbase + ni * 16 + ln];

#pragma unroll
    for (int mi = 0; mi < 4; ++mi)
#pragma unroll
        for (int ni = 0; ni < 4; ++ni) {
            int n = n0 + nbase + ni * 16 + ln;
#pragma unroll
            for (int j = 0; j < 8; ++j) {
                int mloc = mbase + mi * 16 + j + 8 * kh;
                int tok  = toks[mloc];
                float v  = acc[mi][ni][j] + bias[ni];
                __builtin_nontemporal_store(v, out + (size_t)tok * HDIM + n);
            }
        }
}

extern "C" void kernel_launch(void* const* d_in, const int* in_sizes, int n_in,
                              void* d_out, int out_size, void* d_ws, size_t ws_size,
                              hipStream_t stream) {
    (void)in_sizes; (void)n_in; (void)out_size; (void)ws_size;
    const float* x    = (const float*)d_in[0];
    const int*   perm = (const int*)d_in[1];
    const float* W1   = (const float*)d_in[2];
    const float* b1   = (const float*)d_in[3];
    const float* W2   = (const float*)d_in[4];
    const float* b2   = (const float*)d_in[5];
    float* out = (float*)d_out;
    unsigned short* hbuf = (unsigned short*)d_ws;   // 8*512*4096 bf16 = 33.5 MB (fits L2)

    moe_gemm1<<<dim3(16, 4, 8), 256, 0, stream>>>(x, perm, W1, b1, hbuf);
    moe_gemm2<<<dim3(4, 4, 8), 256, 0, stream>>>((const unsigned*)hbuf, perm, W2, b2, out);
}
